// VectorQuantizer_81655918231775
// MI455X (gfx1250) — matrix operand order repaired and checked by reading, not compiled, not run on hardware
//
#include <hip/hip_runtime.h>
#include <math.h>

#define NROWS 32768
#define DDIM  512
#define KCODE 8192
#define BETA  0.25f

// padded+rotated LDS row: 640 elements (1280B) per 512-element row
#define AROW  640

typedef __attribute__((ext_vector_type(16))) __bf16 v16bf;
typedef __attribute__((ext_vector_type(8)))  float  v8f;
typedef __attribute__((ext_vector_type(4)))  unsigned int u32x4;
typedef __attribute__((ext_vector_type(4)))  float  f32x4;

union FragU { v16bf bf; u32x4 q[2]; };

// ---------------------------------------------------------------------------
// Kernel 1: normalize codebook rows, split into bf16 hi/lo (row-major), s_w[k]
// one block (256 thr) per 16-code tile
// ---------------------------------------------------------------------------
__global__ void __launch_bounds__(256) vq_wprep(const float* __restrict__ E,
                                                __bf16* __restrict__ Whi,
                                                __bf16* __restrict__ Wlo,
                                                float* __restrict__ sw) {
  __shared__ float wrow[16 * DDIM];   // 32KB
  __shared__ float part[16][16];
  __shared__ float scls[16];
  const int tile = blockIdx.x;
  const int t = threadIdx.x;
  const size_t gbase = (size_t)tile * 16 * DDIM;
  for (int i = t; i < 16 * DDIM; i += 256) wrow[i] = E[gbase + i];
  __syncthreads();
  {
    const int r = t >> 4, s = t & 15;
    float p = 0.f;
    #pragma unroll 8
    for (int j = 0; j < 32; ++j) { float v = wrow[r * DDIM + s * 32 + j]; p += v * v; }
    part[r][s] = p;
  }
  __syncthreads();
  if (t < 16) {
    float s = 0.f;
    for (int j = 0; j < 16; ++j) s += part[t][j];
    scls[t] = 1.0f / fmaxf(sqrtf(s), 1e-12f);
  }
  __syncthreads();
  {
    const int r = t >> 4, s = t & 15;
    const float scale = scls[r];
    float p = 0.f;
    #pragma unroll 8
    for (int j = 0; j < 32; ++j) {
      int ix = r * DDIM + s * 32 + j;
      float v = wrow[ix] * scale;
      wrow[ix] = v;
      p += v * v;
    }
    part[r][s] = p;
  }
  __syncthreads();
  if (t < 16) {
    float s = 0.f;
    for (int j = 0; j < 16; ++j) s += part[t][j];
    sw[tile * 16 + t] = s;
  }
  for (int i = t; i < 16 * DDIM; i += 256) {
    float v = wrow[i];
    __bf16 hb = (__bf16)v;
    Whi[gbase + i] = hb;
    Wlo[gbase + i] = (__bf16)(v - (float)hb);
  }
}

// ---------------------------------------------------------------------------
// Kernel 2: core. One wave (32 thr) per 16-row strip.
//  - normalize rows + split to bf16 hi/lo into LDS with a rotate+pad layout:
//    row m has stride 640 elems; 16B unit u of row m sits at unit (u + m).
//    -> fragment ds_load_b128 address = per-lane base + 64*c immediate,
//       conflict-free banks (4m..4m+3 per lane), zero VALU addressing.
//  - sweep K in 32-column (2-tile) steps: shared A frags, 6 WMMAs/chunk
//    (hi*hi + hi*lo + lo*hi for each of two column tiles, f32 accum)
//  - unroll 4 (not full): keeps VGPRs < 256 (no s_set_vgpr_msb mode),
//    multi-wave occupancy hides the B stream latency.
//  - per-lane distributed top-3 (C layout: one column per lane), merged with
//    exact (dist, idx) tie-breaking at the end -> indices + inverse-dist weights
// ---------------------------------------------------------------------------
__global__ void __launch_bounds__(32, 2) vq_topk(const float* __restrict__ X,
                                                 const __bf16* __restrict__ Whi,
                                                 const __bf16* __restrict__ Wlo,
                                                 const float* __restrict__ sw,
                                                 unsigned* __restrict__ tki,
                                                 float* __restrict__ tkw,
                                                 float* __restrict__ outIdx) {
  __shared__ alignas(16) __bf16 Ahi[16 * AROW];   // 20KB, rotate+pad layout
  __shared__ alignas(16) __bf16 Alo[16 * AROW];   // 20KB
  __shared__ float red[16][32];
  __shared__ float sxl[16];
  __shared__ float scl[16];
  __shared__ float    mrgd[16][48];
  __shared__ unsigned mrgi[16][48];

  const int l = threadIdx.x;
  const int rowBase = blockIdx.x * 16;
  const float* xb = X + (size_t)rowBase * DDIM;

  // row norms (f32, matches reference normalize with eps)
  for (int m = 0; m < 16; ++m) {
    float p = 0.f;
    #pragma unroll 4
    for (int j = 0; j < 16; ++j) { float v = xb[m * DDIM + l + 32 * j]; p += v * v; }
    red[m][l] = p;
  }
  __syncthreads();
  if (l < 16) {
    float s = 0.f;
    for (int j = 0; j < 32; ++j) s += red[l][j];
    float scale = 1.0f / fmaxf(sqrtf(s), 1e-12f);
    scl[l] = scale;
    sxl[l] = s * scale * scale;   // sum(xn^2)
  }
  __syncthreads();
  // normalize + split; rotated store: element (m,d) -> m*AROW + ((d>>3)+m)*8 + (d&7)
  for (int i = l; i < 16 * DDIM; i += 32) {
    const int m = i >> 9;
    const int d = i & 511;
    const int si = m * AROW + (((d >> 3) + m) << 3) + (d & 7);
    float v = xb[i] * scl[m];
    __bf16 hb = (__bf16)v;
    Ahi[si] = hb;
    Alo[si] = (__bf16)(v - (float)hb);
  }
  __syncthreads();

  const int h = l >> 4;        // half-wave
  const int c16 = l & 15;      // row (A) / column-within-tile (B)
  // per-lane constant A base (elements); chunk c adds 32*c (immediate offsets)
  const int abase = c16 * AROW + ((c16 + 2 * h) << 3);

  float sxr[8];
  #pragma unroll
  for (int r = 0; r < 8; ++r) sxr[r] = sxl[r + h * 8];

  float bd[8][3];
  unsigned bi[8][3];
  #pragma unroll
  for (int r = 0; r < 8; ++r) {
    bd[r][0] = bd[r][1] = bd[r][2] = 3.4e38f;
    bi[r][0] = bi[r][1] = bi[r][2] = 0xFFFFFFFFu;
  }

  for (int tp = 0; tp < KCODE / 32; ++tp) {        // 2 column tiles per step
    v8f acc0 = {};
    v8f acc1 = {};
    const size_t n0 = (size_t)(tp * 32 + c16) * DDIM + h * 16;
    const size_t n1 = n0 + (size_t)16 * DDIM;
    const __bf16* bh0 = Whi + n0;
    const __bf16* bl0 = Wlo + n0;
    const __bf16* bh1 = Whi + n1;
    const __bf16* bl1 = Wlo + n1;
    const unsigned k0 = (unsigned)(tp * 32 + c16);
    const unsigned k1 = k0 + 16u;
    const float sw0 = sw[k0];
    const float sw1 = sw[k1];
    #pragma unroll 4
    for (int c = 0; c < 16; ++c) {
      FragU ah, al, b0h, b0l, b1h, b1l;
      ah.q[0] = *(const u32x4*)&Ahi[abase + c * 32];
      ah.q[1] = *(const u32x4*)&Ahi[abase + c * 32 + 8];
      al.q[0] = *(const u32x4*)&Alo[abase + c * 32];
      al.q[1] = *(const u32x4*)&Alo[abase + c * 32 + 8];
      b0h.q[0] = *(const u32x4*)&bh0[c * 32];
      b0h.q[1] = *(const u32x4*)&bh0[c * 32 + 8];
      b0l.q[0] = *(const u32x4*)&bl0[c * 32];
      b0l.q[1] = *(const u32x4*)&bl0[c * 32 + 8];
      b1h.q[0] = *(const u32x4*)&bh1[c * 32];
      b1h.q[1] = *(const u32x4*)&bh1[c * 32 + 8];
      b1l.q[0] = *(const u32x4*)&bl1[c * 32];
      b1l.q[1] = *(const u32x4*)&bl1[c * 32 + 8];
      acc0 = __builtin_amdgcn_wmma_f32_16x16x32_bf16(false, ah.bf, false, b0h.bf,
                                                     (short)0, acc0, false, false);
      acc0 = __builtin_amdgcn_wmma_f32_16x16x32_bf16(false, ah.bf, false, b0l.bf,
                                                     (short)0, acc0, false, false);
      acc0 = __builtin_amdgcn_wmma_f32_16x16x32_bf16(false, al.bf, false, b0h.bf,
                                                     (short)0, acc0, false, false);
      acc1 = __builtin_amdgcn_wmma_f32_16x16x32_bf16(false, ah.bf, false, b1h.bf,
                                                     (short)0, acc1, false, false);
      acc1 = __builtin_amdgcn_wmma_f32_16x16x32_bf16(false, ah.bf, false, b1l.bf,
                                                     (short)0, acc1, false, false);
      acc1 = __builtin_amdgcn_wmma_f32_16x16x32_bf16(false, al.bf, false, b1h.bf,
                                                     (short)0, acc1, false, false);
    }
    // epilogue: process tile 0 then tile 1 (k strictly increasing per lane,
    // so plain '<' keeps the lower index on ties)
    #pragma unroll
    for (int r = 0; r < 8; ++r) {
      float d = sxr[r] + sw0 - 2.0f * acc0[r];
      if (d < bd[r][0]) {
        bd[r][2] = bd[r][1]; bi[r][2] = bi[r][1];
        bd[r][1] = bd[r][0]; bi[r][1] = bi[r][0];
        bd[r][0] = d;        bi[r][0] = k0;
      } else if (d < bd[r][1]) {
        bd[r][2] = bd[r][1]; bi[r][2] = bi[r][1];
        bd[r][1] = d;        bi[r][1] = k0;
      } else if (d < bd[r][2]) {
        bd[r][2] = d;        bi[r][2] = k0;
      }
    }
    #pragma unroll
    for (int r = 0; r < 8; ++r) {
      float d = sxr[r] + sw1 - 2.0f * acc1[r];
      if (d < bd[r][0]) {
        bd[r][2] = bd[r][1]; bi[r][2] = bi[r][1];
        bd[r][1] = bd[r][0]; bi[r][1] = bi[r][0];
        bd[r][0] = d;        bi[r][0] = k1;
      } else if (d < bd[r][1]) {
        bd[r][2] = bd[r][1]; bi[r][2] = bi[r][1];
        bd[r][1] = d;        bi[r][1] = k1;
      } else if (d < bd[r][2]) {
        bd[r][2] = d;        bi[r][2] = k1;
      }
    }
  }

  // merge: each row's candidates live across its 16 lanes
  #pragma unroll
  for (int r = 0; r < 8; ++r) {
    const int m = r + h * 8;
    #pragma unroll
    for (int s = 0; s < 3; ++s) {
      mrgd[m][c16 * 3 + s] = bd[r][s];
      mrgi[m][c16 * 3 + s] = bi[r][s];
    }
  }
  __syncthreads();
  if (l < 16) {
    float fd[3] = {3.4e38f, 3.4e38f, 3.4e38f};
    unsigned fi[3] = {0xFFFFFFFFu, 0xFFFFFFFFu, 0xFFFFFFFFu};
    for (int j = 0; j < 48; ++j) {
      float d = mrgd[l][j];
      unsigned k = mrgi[l][j];
      if (d < fd[0] || (d == fd[0] && k < fi[0])) {
        fd[2] = fd[1]; fi[2] = fi[1];
        fd[1] = fd[0]; fi[1] = fi[0];
        fd[0] = d;     fi[0] = k;
      } else if (d < fd[1] || (d == fd[1] && k < fi[1])) {
        fd[2] = fd[1]; fi[2] = fi[1];
        fd[1] = d;     fi[1] = k;
      } else if (d < fd[2] || (d == fd[2] && k < fi[2])) {
        fd[2] = d;     fi[2] = k;
      }
    }
    const float i0 = 1.0f / fd[0], i1 = 1.0f / fd[1], i2 = 1.0f / fd[2];
    const float isum = i0 + i1 + i2;
    const int n = rowBase + l;
    tki[n * 3 + 0] = fi[0]; tki[n * 3 + 1] = fi[1]; tki[n * 3 + 2] = fi[2];
    tkw[n * 3 + 0] = i0 / isum; tkw[n * 3 + 1] = i1 / isum; tkw[n * 3 + 2] = i2 / isum;
    outIdx[n * 3 + 0] = (float)fi[0];
    outIdx[n * 3 + 1] = (float)fi[1];
    outIdx[n * 3 + 2] = (float)fi[2];
  }
}

// ---------------------------------------------------------------------------
// Kernel 3: quantize (gather un-normalized codebook rows, weighted sum),
// straight-through output x + (q - x), per-block SSE partial (fixed order).
// 256 threads = 8 waves, one row per wave.
// ---------------------------------------------------------------------------
__global__ void __launch_bounds__(256) vq_quant(const float* __restrict__ X,
                                                const float* __restrict__ E,
                                                const unsigned* __restrict__ tki,
                                                const float* __restrict__ tkw,
                                                float* __restrict__ out,
                                                float* __restrict__ partial) {
  __shared__ float red[256];
  const int n = blockIdx.x * 8 + (threadIdx.x >> 5);
  const int lane = threadIdx.x & 31;
  const unsigned i0 = tki[n * 3 + 0], i1 = tki[n * 3 + 1], i2 = tki[n * 3 + 2];
  const float w0 = tkw[n * 3 + 0], w1 = tkw[n * 3 + 1], w2 = tkw[n * 3 + 2];
  const f32x4* e0 = (const f32x4*)(E + (size_t)i0 * DDIM);
  const f32x4* e1 = (const f32x4*)(E + (size_t)i1 * DDIM);
  const f32x4* e2 = (const f32x4*)(E + (size_t)i2 * DDIM);
  const f32x4* xv = (const f32x4*)(X + (size_t)n * DDIM);
  f32x4* ov = (f32x4*)(out + (size_t)n * DDIM);
  float sse = 0.f;
  #pragma unroll
  for (int j = 0; j < 4; ++j) {
    const int di = j * 32 + lane;
    f32x4 q = w0 * e0[di] + w1 * e1[di] + w2 * e2[di];
    f32x4 x4 = xv[di];
    f32x4 df = q - x4;              // quantized - x (forward of both losses)
    sse += df[0] * df[0] + df[1] * df[1] + df[2] * df[2] + df[3] * df[3];
    ov[di] = x4 + df;               // straight-through: x + (q - x)
  }
  red[threadIdx.x] = sse;
  __syncthreads();
  for (int st = 128; st > 0; st >>= 1) {
    if (threadIdx.x < st) red[threadIdx.x] += red[threadIdx.x + st];
    __syncthreads();
  }
  if (threadIdx.x == 0) partial[blockIdx.x] = red[0];
}

// ---------------------------------------------------------------------------
// Kernel 4: deterministic final loss reduction
// ---------------------------------------------------------------------------
__global__ void __launch_bounds__(256) vq_loss(const float* __restrict__ partial,
                                               float* __restrict__ lossOut) {
  __shared__ float red[256];
  float s = 0.f;
  for (int i = threadIdx.x; i < NROWS / 8; i += 256) s += partial[i];
  red[threadIdx.x] = s;
  __syncthreads();
  for (int st = 128; st > 0; st >>= 1) {
    if (threadIdx.x < st) red[threadIdx.x] += red[threadIdx.x + st];
    __syncthreads();
  }
  if (threadIdx.x == 0)
    lossOut[0] = (1.0f + BETA) * red[0] / (float)((size_t)NROWS * DDIM);
}

// ---------------------------------------------------------------------------
extern "C" void kernel_launch(void* const* d_in, const int* in_sizes, int n_in,
                              void* d_out, int out_size, void* d_ws, size_t ws_size,
                              hipStream_t stream) {
  const float* X = (const float*)d_in[0];        // [N, D]
  const float* E = (const float*)d_in[1];        // [K, D]
  float* out = (float*)d_out;                    // [N*D] quantized_st | [1] loss | [N*3] indices

  char* ws = (char*)d_ws;
  size_t off = 0;
  __bf16* Whi = (__bf16*)(ws + off); off += (size_t)KCODE * DDIM * 2;   // 8 MB
  __bf16* Wlo = (__bf16*)(ws + off); off += (size_t)KCODE * DDIM * 2;   // 8 MB
  float* sw = (float*)(ws + off); off += (size_t)KCODE * 4;
  unsigned* tki = (unsigned*)(ws + off); off += (size_t)NROWS * 3 * 4;
  float* tkw = (float*)(ws + off); off += (size_t)NROWS * 3 * 4;
  float* partial = (float*)(ws + off); off += (size_t)(NROWS / 8) * 4;

  float* qOut = out;
  float* lossOut = out + (size_t)NROWS * DDIM;
  float* idxOut = out + (size_t)NROWS * DDIM + 1;

  vq_wprep<<<KCODE / 16, 256, 0, stream>>>(E, Whi, Wlo, sw);
  vq_topk<<<NROWS / 16, 32, 0, stream>>>(X, Whi, Wlo, sw, tki, tkw, idxOut);
  vq_quant<<<NROWS / 8, 256, 0, stream>>>(X, E, tki, tkw, qOut, partial);
  vq_loss<<<1, 256, 0, stream>>>(partial, lossOut);
}
